// action_graph_17763984736945
// MI455X (gfx1250) — compile-verified, matrix operand order checked by reading, not run on hardware
//
#include <hip/hip_runtime.h>
#include <hip/hip_bf16.h>
#include <math.h>

// ---------------------------------------------------------------------------
// Problem constants (from reference): B=64, R=200, D=512, NUM_STEPS=4
// ---------------------------------------------------------------------------
#define D_DIM   512
#define K_DIM   512
#define N_DIM   1536          // 3*D
#define R_DIM   200
#define B_DIM   64
#define MV      (B_DIM * 201) // 12864 rows (vert sequences)
#define ME      (B_DIM * 200) // 12800 rows (edge sequences)
#define NSTEPS  4

// GEMM tiling: block tile 128x128, K-chunk 32; 8 waves, each wave 32(M)x64(N)
#define BM 128
#define BN 128
#define BK 32
#define BKP 40                // padded LDS row stride (bf16 elems), keeps 16B align

typedef __attribute__((ext_vector_type(16))) __bf16 v16bf;
typedef __attribute__((ext_vector_type(8)))  __bf16 v8bf;
typedef __attribute__((ext_vector_type(8)))  float  v8f;

// ---------------------------------------------------------------------------
// GEMM: Y[M x 1536] = X[M x 512] * W[1536 x 512]^T + bias, via bf16 WMMA
// ---------------------------------------------------------------------------
__global__ __launch_bounds__(256)
void gemm_bias_wmma(const float* __restrict__ X, const float* __restrict__ W,
                    const float* __restrict__ bias, float* __restrict__ Y, int M)
{
    __shared__ __bf16 As[BM * BKP];
    __shared__ __bf16 Bs[BN * BKP];

    const int tid  = threadIdx.x;
    const int m0   = blockIdx.x * BM;
    const int n0   = blockIdx.y * BN;
    const int wave = tid >> 5;
    const int lane = tid & 31;
    const int wm   = (wave & 3) * 32;   // wave row offset inside block tile
    const int wn   = (wave >> 2) * 64;  // wave col offset inside block tile

    v8f acc[2][4] = {};

    // Staging: both tiles are 128 rows x 32 cols; 2 threads per row, 16 f32 each.
    const int srow  = tid >> 1;
    const int skoff = (tid & 1) * 16;

    for (int kc = 0; kc < K_DIM; kc += BK) {
        // ---- stage A tile (BM x BK), f32 -> bf16 (zero-fill past M) ----
        {
            float tmp[16];
            const int grow = m0 + srow;
            if (grow < M) {
                const float4* src = (const float4*)(X + (size_t)grow * K_DIM + kc + skoff);
                #pragma unroll
                for (int i = 0; i < 4; ++i) {
                    float4 v = src[i];
                    tmp[4*i] = v.x; tmp[4*i+1] = v.y; tmp[4*i+2] = v.z; tmp[4*i+3] = v.w;
                }
            } else {
                #pragma unroll
                for (int i = 0; i < 16; ++i) tmp[i] = 0.f;
            }
            __bf16* dst = &As[srow * BKP + skoff];
            #pragma unroll
            for (int i = 0; i < 16; ++i) dst[i] = (__bf16)tmp[i];
        }
        // ---- stage B tile (BN x BK): Bs[n][k] = W[n0+n][kc+k] ----
        {
            float tmp[16];
            const float4* src = (const float4*)(W + (size_t)(n0 + srow) * K_DIM + kc + skoff);
            #pragma unroll
            for (int i = 0; i < 4; ++i) {
                float4 v = src[i];
                tmp[4*i] = v.x; tmp[4*i+1] = v.y; tmp[4*i+2] = v.z; tmp[4*i+3] = v.w;
            }
            __bf16* dst = &Bs[srow * BKP + skoff];
            #pragma unroll
            for (int i = 0; i < 16; ++i) dst[i] = (__bf16)tmp[i];
        }
        __syncthreads();

        // ---- fragment loads per ISA 16-bit A layout:
        //   lanes 0-15: row, K {0..7,16..23}; lanes 16-31: same row, K {8..15,24..31}
        //   (B mirrored with lane&15 = output column = row of W)
        const int half = (lane >> 4) * 8;
        const int lrow = lane & 15;
        v16bf afrag[2], bfrag[4];
        #pragma unroll
        for (int t = 0; t < 2; ++t) {
            const __bf16* pa = &As[(wm + t * 16 + lrow) * BKP + half];
            v8bf alo = *(const v8bf*)(pa);
            v8bf ahi = *(const v8bf*)(pa + 16);
            #pragma unroll
            for (int i = 0; i < 8; ++i) { afrag[t][i] = alo[i]; afrag[t][8 + i] = ahi[i]; }
        }
        #pragma unroll
        for (int t = 0; t < 4; ++t) {
            const __bf16* pb = &Bs[(wn + t * 16 + lrow) * BKP + half];
            v8bf blo = *(const v8bf*)(pb);
            v8bf bhi = *(const v8bf*)(pb + 16);
            #pragma unroll
            for (int i = 0; i < 8; ++i) { bfrag[t][i] = blo[i]; bfrag[t][8 + i] = bhi[i]; }
        }
        #pragma unroll
        for (int tm = 0; tm < 2; ++tm)
            #pragma unroll
            for (int tn = 0; tn < 4; ++tn)
                acc[tm][tn] = __builtin_amdgcn_wmma_f32_16x16x32_bf16(
                    false, afrag[tm], false, bfrag[tn],
                    (short)0, acc[tm][tn], false, false);
        __syncthreads();
    }

    // ---- epilogue: C layout (lane&15 = N, VGPR i -> M = i | 8+i) + bias ----
    const int lrow = lane & 15;
    const int rsel = (lane >> 4) * 8;
    #pragma unroll
    for (int tm = 0; tm < 2; ++tm) {
        #pragma unroll
        for (int tn = 0; tn < 4; ++tn) {
            const int   col     = n0 + wn + tn * 16 + lrow;
            const float bv      = bias[col];
            const int   rowBase = m0 + wm + tm * 16 + rsel;
            #pragma unroll
            for (int i = 0; i < 8; ++i) {
                const int row = rowBase + i;
                if (row < M) Y[(size_t)row * N_DIM + col] = acc[tm][tn][i] + bv;
            }
        }
    }
}

// ---------------------------------------------------------------------------
// Fused GRU gate combine:  out = (1-z)*n + z*h   (gh==null => h=0, gh=bhh)
// ---------------------------------------------------------------------------
__global__ __launch_bounds__(256)
void gru_combine_kernel(const float* __restrict__ gi, const float* __restrict__ gh,
                        const float* __restrict__ h,  const float* __restrict__ bhh,
                        float* __restrict__ out, int M)
{
    const size_t idx = (size_t)blockIdx.x * blockDim.x + threadIdx.x;
    if (idx >= (size_t)M * D_DIM) return;
    const int    d  = (int)(idx & (D_DIM - 1));
    const size_t m  = idx >> 9;
    const size_t gb = m * N_DIM;
    const float ir = gi[gb + d], iz = gi[gb + D_DIM + d], in = gi[gb + 2 * D_DIM + d];
    float hr, hz, hn, hv;
    if (gh) { hr = gh[gb + d]; hz = gh[gb + D_DIM + d]; hn = gh[gb + 2 * D_DIM + d]; hv = h[idx]; }
    else    { hr = bhh[d];     hz = bhh[D_DIM + d];     hn = bhh[2 * D_DIM + d];     hv = 0.f;    }
    const float rg = 1.f / (1.f + __expf(-(ir + hr)));
    const float zg = 1.f / (1.f + __expf(-(iz + hz)));
    const float ng = tanhf(in + rg * hn);
    out[idx] = (1.f - zg) * ng + zg * hv;
}

// ---------------------------------------------------------------------------
// Attention raw scores: per (b,r) three tanh'd dot products over D=512
//   s1 = tanh(verb.eWa + edge.eWb + eb)     (att(verb_concat, edge_att))
//   s2 = tanh(region.eWa + edge.eWb + eb)   (att(region_concat, edge_att))
//   s3 = tanh(verb.vWa + edge.vWb + vb)     (att(verb_concat, vert_att))
// ---------------------------------------------------------------------------
__global__ __launch_bounds__(256)
void att_scores_kernel(const float* __restrict__ vert, const float* __restrict__ edge,
                       const float* __restrict__ eW, const float* __restrict__ eb,
                       const float* __restrict__ vW, const float* __restrict__ vb,
                       float* __restrict__ s1, float* __restrict__ s2, float* __restrict__ s3)
{
    const int r = blockIdx.x, b = blockIdx.y, tid = threadIdx.x;
    const float* verb = vert + (size_t)(b * 201) * D_DIM;
    const float* reg  = vert + (size_t)(b * 201 + 1 + r) * D_DIM;
    const float* edg  = edge + (size_t)(b * 200 + r) * D_DIM;

    float p[5] = {0.f, 0.f, 0.f, 0.f, 0.f};
    for (int d = tid; d < D_DIM; d += 256) {
        const float vv = verb[d], rg = reg[d], ed = edg[d];
        const float ea = eW[d], eb2 = eW[D_DIM + d];
        const float va = vW[d], vb2 = vW[D_DIM + d];
        p[0] += vv * ea; p[1] += ed * eb2; p[2] += rg * ea;
        p[3] += vv * va; p[4] += ed * vb2;
    }
    __shared__ float red[5][8];
    #pragma unroll
    for (int i = 0; i < 5; ++i) {
        float v = p[i];
        for (int off = 16; off > 0; off >>= 1) v += __shfl_xor(v, off, 32);
        if ((tid & 31) == 0) red[i][tid >> 5] = v;
    }
    __syncthreads();
    if (tid == 0) {
        float t[5];
        #pragma unroll
        for (int i = 0; i < 5; ++i) {
            float a = 0.f;
            for (int w = 0; w < 8; ++w) a += red[i][w];
            t[i] = a;
        }
        const float ebb = eb[0], vbb = vb[0];
        const int o = b * R_DIM + r;
        s1[o] = tanhf(t[0] + t[1] + ebb);
        s2[o] = tanhf(t[2] + t[1] + ebb);
        s3[o] = tanhf(t[3] + t[4] + vbb);
    }
}

// ---------------------------------------------------------------------------
// In-place log_softmax over r (R=200) for each of the three score arrays
// ---------------------------------------------------------------------------
__global__ __launch_bounds__(256)
void logsoftmax_kernel(float* __restrict__ s1, float* __restrict__ s2, float* __restrict__ s3)
{
    const int b = blockIdx.x, tid = threadIdx.x;
    float* s = (blockIdx.y == 0) ? s1 : (blockIdx.y == 1 ? s2 : s3);
    s += (size_t)b * R_DIM;

    const float v = (tid < R_DIM) ? s[tid] : -3.4e38f;
    __shared__ float sm[8];
    __shared__ float bmax, bsum;

    float m = v;
    for (int off = 16; off > 0; off >>= 1) m = fmaxf(m, __shfl_xor(m, off, 32));
    if ((tid & 31) == 0) sm[tid >> 5] = m;
    __syncthreads();
    if (tid == 0) { float mm = sm[0]; for (int i = 1; i < 8; ++i) mm = fmaxf(mm, sm[i]); bmax = mm; }
    __syncthreads();

    float e = (tid < R_DIM) ? __expf(v - bmax) : 0.f;
    for (int off = 16; off > 0; off >>= 1) e += __shfl_xor(e, off, 32);
    if ((tid & 31) == 0) sm[tid >> 5] = e;
    __syncthreads();
    if (tid == 0) { float ss = 0.f; for (int i = 0; i < 8; ++i) ss += sm[i]; bsum = ss; }
    __syncthreads();

    if (tid < R_DIM) s[tid] = v - bmax - __logf(bsum);
}

// ---------------------------------------------------------------------------
// edge_ctx[b,r,:] = s1*verb + s2*region ; vert_ctx[b,1+r,:] = s2*edge
// ---------------------------------------------------------------------------
__global__ __launch_bounds__(256)
void build_ctx_kernel(const float* __restrict__ vert, const float* __restrict__ edge,
                      const float* __restrict__ s1, const float* __restrict__ s2,
                      float* __restrict__ edge_ctx, float* __restrict__ vert_ctx)
{
    const size_t idx = (size_t)blockIdx.x * blockDim.x + threadIdx.x;
    if (idx >= (size_t)ME * D_DIM) return;
    const int d = (int)(idx & (D_DIM - 1));
    const int t = (int)(idx >> 9);          // b*200 + r
    const int r = t % R_DIM;
    const int b = t / R_DIM;
    const float w1 = s1[t], w2 = s2[t];
    const float verbv = vert[((size_t)b * 201) * D_DIM + d];
    const float regv  = vert[((size_t)(b * 201 + 1 + r)) * D_DIM + d];
    edge_ctx[idx] = w1 * verbv + w2 * regv;
    vert_ctx[((size_t)(b * 201 + 1 + r)) * D_DIM + d] = w2 * edge[idx];
}

// ---------------------------------------------------------------------------
// vert_ctx[b,0,:] = sum_r s3[b,r] * edge[b,r,:]
// ---------------------------------------------------------------------------
__global__ __launch_bounds__(256)
void aw_verb_kernel(const float* __restrict__ edge, const float* __restrict__ s3,
                    float* __restrict__ vert_ctx)
{
    const int b = blockIdx.x;
    __shared__ float w[R_DIM];
    for (int i = threadIdx.x; i < R_DIM; i += 256) w[i] = s3[b * R_DIM + i];
    __syncthreads();
    for (int d = threadIdx.x; d < D_DIM; d += 256) {
        float acc = 0.f;
        const float* base = edge + (size_t)b * R_DIM * D_DIM + d;
        for (int r = 0; r < R_DIM; ++r) acc += w[r] * base[(size_t)r * D_DIM];
        vert_ctx[(size_t)b * 201 * D_DIM + d] = acc;
    }
}

// ---------------------------------------------------------------------------
// Host orchestration
// ---------------------------------------------------------------------------
static void launch_gemm(const float* X, const float* W, const float* bias,
                        float* Y, int M, hipStream_t stream)
{
    dim3 grid((M + BM - 1) / BM, N_DIM / BN);   // N: 1536/128 = 12 exact
    gemm_bias_wmma<<<grid, 256, 0, stream>>>(X, W, bias, Y, M);
}

extern "C" void kernel_launch(void* const* d_in, const int* in_sizes, int n_in,
                              void* d_out, int out_size, void* d_ws, size_t ws_size,
                              hipStream_t stream)
{
    (void)in_sizes; (void)n_in; (void)out_size; (void)ws_size;
    const float* input0 = (const float*)d_in[0];
    const float* input1 = (const float*)d_in[1];
    const float* vWih   = (const float*)d_in[2];
    const float* vWhh   = (const float*)d_in[3];
    const float* vbih   = (const float*)d_in[4];
    const float* vbhh   = (const float*)d_in[5];
    const float* eWih   = (const float*)d_in[6];
    const float* eWhh   = (const float*)d_in[7];
    const float* ebih   = (const float*)d_in[8];
    const float* ebhh   = (const float*)d_in[9];
    const float* eAttW  = (const float*)d_in[10];
    const float* eAttB  = (const float*)d_in[11];
    const float* vAttW  = (const float*)d_in[12];
    const float* vAttB  = (const float*)d_in[13];

    // States live directly in d_out (fully overwritten by initial GRUs).
    float* vert_state = (float*)d_out;                       // MV x 512
    float* edge_state = vert_state + (size_t)MV * D_DIM;     // ME x 512

    float* ws       = (float*)d_ws;
    float* vert_ctx = ws;                                    // MV x 512
    float* edge_ctx = vert_ctx + (size_t)MV * D_DIM;         // ME x 512
    float* gi       = edge_ctx + (size_t)ME * D_DIM;         // MV x 1536
    float* gh       = gi + (size_t)MV * N_DIM;               // MV x 1536
    float* s1       = gh + (size_t)MV * N_DIM;               // 64*200
    float* s2       = s1 + ME;
    float* s3       = s2 + ME;

    const int gruV = (int)(((size_t)MV * D_DIM + 255) / 256);
    const int gruE = (int)(((size_t)ME * D_DIM + 255) / 256);

    // ---- initial GRUs (h = 0 -> skip Whh GEMM; gh degenerates to bhh) ----
    launch_gemm(input0, vWih, vbih, gi, MV, stream);
    gru_combine_kernel<<<gruV, 256, 0, stream>>>(gi, nullptr, nullptr, vbhh, vert_state, MV);
    launch_gemm(input1, eWih, ebih, gi, ME, stream);
    gru_combine_kernel<<<gruE, 256, 0, stream>>>(gi, nullptr, nullptr, ebhh, edge_state, ME);

    for (int step = 0; step < NSTEPS; ++step) {
        att_scores_kernel<<<dim3(R_DIM, B_DIM), 256, 0, stream>>>(
            vert_state, edge_state, eAttW, eAttB, vAttW, vAttB, s1, s2, s3);
        logsoftmax_kernel<<<dim3(B_DIM, 3), 256, 0, stream>>>(s1, s2, s3);
        build_ctx_kernel<<<gruE, 256, 0, stream>>>(vert_state, edge_state, s1, s2,
                                                   edge_ctx, vert_ctx);
        aw_verb_kernel<<<B_DIM, 256, 0, stream>>>(edge_state, s3, vert_ctx);

        // edge GRU: edge_state = GRU(edge_ctx, edge_state)
        launch_gemm(edge_ctx, eWih, ebih, gi, ME, stream);
        launch_gemm(edge_state, eWhh, ebhh, gh, ME, stream);
        gru_combine_kernel<<<gruE, 256, 0, stream>>>(gi, gh, edge_state, ebhh, edge_state, ME);

        // vert GRU: vert_state = GRU(vert_ctx, vert_state)   (old vert_state as h)
        launch_gemm(vert_ctx, vWih, vbih, gi, MV, stream);
        launch_gemm(vert_state, vWhh, vbhh, gh, MV, stream);
        gru_combine_kernel<<<gruV, 256, 0, stream>>>(gi, gh, vert_state, vbhh, vert_state, MV);
    }
}
